// Model_16484084482977
// MI455X (gfx1250) — compile-verified, matrix operand order checked by reading, not run on hardware
//
#include <hip/hip_runtime.h>
#include <stdint.h>

typedef __attribute__((ext_vector_type(16))) _Float16 v16h;
typedef __attribute__((ext_vector_type(8)))  float    v8f;

#define HDIM 128
#define N_IS0 262144
#define N_ID0 65536
#define N_ID1 8192
#define N_UD0 65536
#define N_UD1 8192
#define E0 655360
#define E1 81920
#define EP 16384

union FragA {
  v16h h;
  uint32_t u[8];
};

// ---------------- gather + f32->f16 convert ----------------
__global__ __launch_bounds__(256) void k_gather_h(
    const float* __restrict__ table, const int* __restrict__ ids,
    _Float16* __restrict__ out, int rows)
{
  int t = blockIdx.x * 256 + threadIdx.x;
  int r = t >> 7, c = t & 127;
  if (r >= rows) return;
  out[(size_t)r * HDIM + c] = (_Float16)table[(size_t)ids[r] * HDIM + c];
}

// ---------------- edge scatter-add (mean aggregation numerator + degree) ----
// One wave per edge; accumulators (<=32MB) are L2-resident so f32 global
// atomics are cheap. Source rows come either from a f32 table via an id map
// (block0: item_emb[item_src_ids[src]]) or from a f16 activation matrix.
__global__ __launch_bounds__(256) void k_agg(
    const int* __restrict__ esrc, const int* __restrict__ edst, int nE,
    const float* __restrict__ srcF, const int* __restrict__ srcmap,
    const _Float16* __restrict__ srcH,
    float* __restrict__ agg, float* __restrict__ deg)
{
  int e = blockIdx.x * 8 + (threadIdx.x >> 5);
  if (e >= nE) return;
  int lane = threadIdx.x & 31;
  int s = esrc[e];
  int d = edst[e];
  if (lane == 0) atomicAdd(&deg[d], 1.0f);
  if (srcF) {
    size_t r = (size_t)srcmap[s];
    #pragma unroll
    for (int j = 0; j < 4; ++j) {
      int c = lane + 32 * j;
      atomicAdd(&agg[(size_t)d * HDIM + c], srcF[r * HDIM + c]);
    }
  } else {
    #pragma unroll
    for (int j = 0; j < 4; ++j) {
      int c = lane + 32 * j;
      atomicAdd(&agg[(size_t)d * HDIM + c], (float)srcH[(size_t)s * HDIM + c]);
    }
  }
}

// ---------------- mean = agg / max(deg,1), converted to f16 ----------------
__global__ __launch_bounds__(256) void k_mean_h(
    const float* __restrict__ agg, const float* __restrict__ deg,
    _Float16* __restrict__ out, int rows)
{
  int t = blockIdx.x * 256 + threadIdx.x;
  int r = t >> 7, c = t & 127;
  if (r >= rows) return;
  float dv = fmaxf(deg[r], 1.0f);
  out[(size_t)r * HDIM + c] = (_Float16)(agg[(size_t)r * HDIM + c] / dv);
}

// ---------------- concat two int index arrays ----------------
__global__ __launch_bounds__(256) void k_concat_idx(
    const int* __restrict__ a, const int* __restrict__ b,
    int* __restrict__ out, int n)
{
  int t = blockIdx.x * 256 + threadIdx.x;
  if (t < n) out[t] = a[t];
  else if (t < 2 * n) out[t] = b[t - n];
}

// ---------------- fused dual GEMM via WMMA --------------------------------
// Y[M,128] = act( Xs[rowS(m)] @ Ws  (+ Xn[rowN(m)] @ Wn)  + bias )
// Grid: M/128 workgroups of 256 threads (8 waves). Wave w owns rows
// [blk*128 + 16w, +16) x all 128 cols (8 16x16 C tiles, v8f each).
// Weights are transposed into LDS as f16 so each B fragment is two
// contiguous ds_load_b128s; A fragments are two global_load_b128s.
// Templated so the inner WMMA chain is branch-free.
template <bool NEIGH, bool IDX, bool RELU>
__global__ __launch_bounds__(256) void k_sage_gemm(
    const _Float16* __restrict__ Xs, const int* __restrict__ idxS,
    const _Float16* __restrict__ Xn, const int* __restrict__ idxN,
    const float* __restrict__ Ws, const float* __restrict__ Wn,
    const float* __restrict__ bias, _Float16* __restrict__ Yh)
{
  __shared__ _Float16 wsT[HDIM * HDIM];
  __shared__ _Float16 wnT[NEIGH ? HDIM * HDIM : 16];
  for (int i = threadIdx.x; i < HDIM * HDIM; i += 256) {
    int k = i >> 7, n = i & 127;
    wsT[n * HDIM + k] = (_Float16)Ws[i];
    if (NEIGH) wnT[n * HDIM + k] = (_Float16)Wn[i];
  }
  __syncthreads();

  const int wave = threadIdx.x >> 5;
  const int lane = threadIdx.x & 31;
  const int m0 = blockIdx.x * 128 + wave * 16;
  const int mrow = m0 + (lane & 15);
  const size_t rowS = IDX ? (size_t)idxS[mrow] : (size_t)mrow;
  const size_t rowN = NEIGH ? (IDX ? (size_t)idxN[mrow] : (size_t)mrow) : 0;
  const int klane = (lane >= 16) ? 8 : 0;   // upper half-wave holds K+8 pairs
  const int ncol0 = lane & 15;

  v8f acc[8] = {};

  #pragma unroll
  for (int kc = 0; kc < 4; ++kc) {
    FragA aS = {}, aN = {};
    #pragma unroll
    for (int r = 0; r < 8; ++r) {
      int k = kc * 32 + ((r < 4) ? 0 : 16) + klane + 2 * (r & 3);
      aS.u[r] = *(const uint32_t*)(Xs + rowS * HDIM + k);
      if (NEIGH) aN.u[r] = *(const uint32_t*)(Xn + rowN * HDIM + k);
    }
    #pragma unroll
    for (int nt = 0; nt < 8; ++nt) {
      int ncol = nt * 16 + ncol0;
      FragA b = {};
      #pragma unroll
      for (int r = 0; r < 8; ++r) {
        int k = kc * 32 + ((r < 4) ? 0 : 16) + klane + 2 * (r & 3);
        b.u[r] = *(const uint32_t*)&wsT[ncol * HDIM + k];
      }
      acc[nt] = __builtin_amdgcn_wmma_f32_16x16x32_f16(
          false, aS.h, false, b.h, (short)0, acc[nt], false, false);
      if (NEIGH) {
        FragA bn = {};
        #pragma unroll
        for (int r = 0; r < 8; ++r) {
          int k = kc * 32 + ((r < 4) ? 0 : 16) + klane + 2 * (r & 3);
          bn.u[r] = *(const uint32_t*)&wnT[ncol * HDIM + k];
        }
        acc[nt] = __builtin_amdgcn_wmma_f32_16x16x32_f16(
            false, aN.h, false, bn.h, (short)0, acc[nt], false, false);
      }
    }
  }

  // C/D layout: VGPR v -> row m0 + (lane>=16 ? 8 : 0) + v, col = nt*16 + lane%16
  const int mbase = m0 + ((lane >= 16) ? 8 : 0);
  #pragma unroll
  for (int nt = 0; nt < 8; ++nt) {
    int n = nt * 16 + ncol0;
    float bv = bias[n];
    #pragma unroll
    for (int v = 0; v < 8; ++v) {
      float val = acc[nt][v] + bv;
      if (RELU) val = fmaxf(val, 0.0f);
      Yh[(size_t)(mbase + v) * HDIM + n] = (_Float16)val;
    }
  }
}

// ---------------- final [128]->[1] projection (wave reduction) ------------
__global__ __launch_bounds__(256) void k_dot_out(
    const _Float16* __restrict__ h1, const float* __restrict__ w,
    const float* __restrict__ b, float* __restrict__ out, int M)
{
  int e = blockIdx.x * 8 + (threadIdx.x >> 5);
  if (e >= M) return;
  int lane = threadIdx.x & 31;
  float s = 0.0f;
  #pragma unroll
  for (int j = 0; j < 4; ++j) {
    int c = lane + 32 * j;
    s += (float)h1[(size_t)e * HDIM + c] * w[c];
  }
  #pragma unroll
  for (int off = 16; off > 0; off >>= 1) s += __shfl_down(s, off, 32);
  if (lane == 0) out[e] = s + b[0];
}

extern "C" void kernel_launch(void* const* d_in, const int* in_sizes, int n_in,
                              void* d_out, int out_size, void* d_ws, size_t ws_size,
                              hipStream_t stream)
{
  const float* user_emb = (const float*)d_in[0];
  const float* item_emb = (const float*)d_in[1];
  const float* iw1s = (const float*)d_in[2];
  const float* iw1n = (const float*)d_in[3];
  const float* ib1  = (const float*)d_in[4];
  const float* iw2s = (const float*)d_in[5];
  const float* iw2n = (const float*)d_in[6];
  const float* ib2  = (const float*)d_in[7];
  const float* uw1s = (const float*)d_in[8];
  const float* uw1n = (const float*)d_in[9];
  const float* ub1  = (const float*)d_in[10];
  const float* uw2s = (const float*)d_in[11];
  const float* uw2n = (const float*)d_in[12];
  const float* ub2  = (const float*)d_in[13];
  const float* uw3s = (const float*)d_in[14];
  const float* uw3n = (const float*)d_in[15];
  const float* ub3  = (const float*)d_in[16];
  const float* ulw  = (const float*)d_in[17];
  const float* ulb  = (const float*)d_in[18];
  const float* dw1  = (const float*)d_in[19];
  const float* db1  = (const float*)d_in[20];
  const float* dw2  = (const float*)d_in[21];
  const float* db2  = (const float*)d_in[22];
  const int* item_src_ids = (const int*)d_in[23];
  const int* user_dst_ids = (const int*)d_in[24];
  const int* e0ii_src = (const int*)d_in[25];
  const int* e0ii_dst = (const int*)d_in[26];
  const int* e0iu_src = (const int*)d_in[27];
  const int* e0iu_dst = (const int*)d_in[28];
  const int* e1ii_src = (const int*)d_in[29];
  const int* e1ii_dst = (const int*)d_in[30];
  const int* e1iu_src = (const int*)d_in[31];
  const int* e1iu_dst = (const int*)d_in[32];
  const int* pos_src  = (const int*)d_in[33];
  const int* pos_dst  = (const int*)d_in[34];
  const int* neg_src  = (const int*)d_in[35];
  const int* neg_dst  = (const int*)d_in[36];
  (void)in_sizes; (void)n_in; (void)out_size; (void)ws_size;

  // ---- carve workspace ----
  char* p = (char*)d_ws;
  auto carve = [&](size_t bytes) -> void* {
    void* q = (void*)p;
    p += (bytes + 255) & ~(size_t)255;
    return q;
  };
  _Float16* xi_h  = (_Float16*)carve((size_t)N_ID0 * HDIM * 2); // xi[:N_ID0] f16
  _Float16* xu_h  = (_Float16*)carve((size_t)N_UD1 * HDIM * 2); // xu[:N_UD1] f16
  float*    agg   = (float*)   carve((size_t)N_ID0 * HDIM * 4); // reused 4x
  float*    deg   = (float*)   carve((size_t)N_ID0 * 4);        // reused 4x
  _Float16* m0ii  = (_Float16*)carve((size_t)N_ID0 * HDIM * 2);
  _Float16* m0iu  = (_Float16*)carve((size_t)N_UD1 * HDIM * 2);
  _Float16* i1_h  = (_Float16*)carve((size_t)N_ID0 * HDIM * 2);
  _Float16* ui1_h = (_Float16*)carve((size_t)N_ID0 * HDIM * 2);
  _Float16* u1_h  = (_Float16*)carve((size_t)N_UD1 * HDIM * 2);
  _Float16* m1ii  = (_Float16*)carve((size_t)N_ID1 * HDIM * 2);
  _Float16* m1iu  = (_Float16*)carve((size_t)N_UD1 * HDIM * 2);
  _Float16* u2_h  = (_Float16*)carve((size_t)N_UD1 * HDIM * 2);
  _Float16* zi_h  = (_Float16*)carve((size_t)N_ID1 * HDIM * 2);
  _Float16* zu_h  = (_Float16*)carve((size_t)N_UD1 * HDIM * 2);
  int*      scat  = (int*)     carve((size_t)2 * EP * 4);
  int*      dcat  = (int*)     carve((size_t)2 * EP * 4);
  _Float16* h1_h  = (_Float16*)carve((size_t)2 * EP * HDIM * 2);

  // ---- gathers (only the rows later consumed as GEMM "self" inputs) ----
  k_gather_h<<<(N_ID0 * HDIM) / 256, 256, 0, stream>>>(item_emb, item_src_ids, xi_h, N_ID0);
  k_gather_h<<<(N_UD1 * HDIM) / 256, 256, 0, stream>>>(user_emb, user_dst_ids, xu_h, N_UD1);

  // ---- block0 item-item mean (shared by item & user encoders) ----
  hipMemsetAsync(agg, 0, (size_t)N_ID0 * HDIM * 4, stream);
  hipMemsetAsync(deg, 0, (size_t)N_ID0 * 4, stream);
  k_agg<<<E0 / 8, 256, 0, stream>>>(e0ii_src, e0ii_dst, E0,
                                    item_emb, item_src_ids, nullptr, agg, deg);
  k_mean_h<<<(N_ID0 * HDIM) / 256, 256, 0, stream>>>(agg, deg, m0ii, N_ID0);

  // ---- block0 item->user mean (only first N_UD1 rows needed later) ----
  hipMemsetAsync(agg, 0, (size_t)N_UD0 * HDIM * 4, stream);
  hipMemsetAsync(deg, 0, (size_t)N_UD0 * 4, stream);
  k_agg<<<E0 / 8, 256, 0, stream>>>(e0iu_src, e0iu_dst, E0,
                                    item_emb, item_src_ids, nullptr, agg, deg);
  k_mean_h<<<(N_UD1 * HDIM) / 256, 256, 0, stream>>>(agg, deg, m0iu, N_UD1);

  // ---- layer-1 GEMMs ----
  k_sage_gemm<true, false, true><<<N_ID0 / 128, 256, 0, stream>>>(
      xi_h, nullptr, m0ii, nullptr, iw1s, iw1n, ib1, i1_h);
  k_sage_gemm<true, false, true><<<N_ID0 / 128, 256, 0, stream>>>(
      xi_h, nullptr, m0ii, nullptr, uw1s, uw1n, ub1, ui1_h);
  k_sage_gemm<true, false, true><<<N_UD1 / 128, 256, 0, stream>>>(
      xu_h, nullptr, m0iu, nullptr, uw2s, uw2n, ub2, u1_h);

  // ---- block1 item-item: z_item ----
  hipMemsetAsync(agg, 0, (size_t)N_ID1 * HDIM * 4, stream);
  hipMemsetAsync(deg, 0, (size_t)N_ID1 * 4, stream);
  k_agg<<<E1 / 8, 256, 0, stream>>>(e1ii_src, e1ii_dst, E1,
                                    nullptr, nullptr, i1_h, agg, deg);
  k_mean_h<<<(N_ID1 * HDIM) / 256, 256, 0, stream>>>(agg, deg, m1ii, N_ID1);
  k_sage_gemm<true, false, true><<<N_ID1 / 128, 256, 0, stream>>>(
      i1_h, nullptr, m1ii, nullptr, iw2s, iw2n, ib2, zi_h);

  // ---- block1 item->user: u2, then z_user ----
  hipMemsetAsync(agg, 0, (size_t)N_UD1 * HDIM * 4, stream);
  hipMemsetAsync(deg, 0, (size_t)N_UD1 * 4, stream);
  k_agg<<<E1 / 8, 256, 0, stream>>>(e1iu_src, e1iu_dst, E1,
                                    nullptr, nullptr, ui1_h, agg, deg);
  k_mean_h<<<(N_UD1 * HDIM) / 256, 256, 0, stream>>>(agg, deg, m1iu, N_UD1);
  k_sage_gemm<true, false, true><<<N_UD1 / 128, 256, 0, stream>>>(
      u1_h, nullptr, m1iu, nullptr, uw3s, uw3n, ub3, u2_h);
  k_sage_gemm<false, false, false><<<N_UD1 / 128, 256, 0, stream>>>(
      u2_h, nullptr, nullptr, nullptr, ulw, nullptr, ulb, zu_h);

  // ---- decoder: concat-gather dual GEMM + relu, then dot with dw2 ----
  k_concat_idx<<<(2 * EP) / 256, 256, 0, stream>>>(pos_src, neg_src, scat, EP);
  k_concat_idx<<<(2 * EP) / 256, 256, 0, stream>>>(pos_dst, neg_dst, dcat, EP);
  k_sage_gemm<true, true, true><<<(2 * EP) / 128, 256, 0, stream>>>(
      zu_h, scat, zi_h, dcat, dw1, dw1 + HDIM * HDIM, db1, h1_h);
  k_dot_out<<<(2 * EP) / 8, 256, 0, stream>>>(h1_h, dw2, db2, (float*)d_out, 2 * EP);
}
